// MultiHeadAttention_25271587569823
// MI455X (gfx1250) — matrix operand order repaired and checked by reading, not compiled, not run on hardware
//
#include <hip/hip_runtime.h>

// ---------------------------------------------------------------------------
// MHA-without-softmax as a 4-GEMM chain on CDNA5 WMMA (bf16 in, f32 accum).
// B=4, SEQ=DM=2048, N_HEADS=1 -> head_dim = 2048.
// All GEMMs are NT (B pre-transposed to [N x K]); double-buffered LDS pipeline
// fed by async global->LDS DMA (ASYNCcnt) when the toolchain exposes it.
// ---------------------------------------------------------------------------

typedef __attribute__((ext_vector_type(16))) __bf16 v16bf;
typedef __attribute__((ext_vector_type(8)))  __bf16 v8bf;
typedef __attribute__((ext_vector_type(8)))  float  v8f;
typedef __attribute__((ext_vector_type(4)))  int    v4i;

typedef __attribute__((address_space(1))) v4i gv4i;  // global 16B chunk
typedef __attribute__((address_space(3))) v4i sv4i;  // LDS 16B chunk

#define BLK_M 128
#define BLK_N 256
#define BLK_K 32
#define LDSP  40                 // 32 + 8 pad: 16B-aligned rows, conflict-free b128
#define ASZ   (BLK_M * LDSP)
#define BSZ   (BLK_N * LDSP)

#if __has_builtin(__builtin_amdgcn_global_load_async_to_lds_b128)
#define HAS_ASYNC_LDS 1
#else
#define HAS_ASYNC_LDS 0
#endif

__device__ __forceinline__ __bf16 f2bf(float f) {
  unsigned int u = __builtin_bit_cast(unsigned int, f);
  u += 0x7FFFu + ((u >> 16) & 1u);
  unsigned short s = (unsigned short)(u >> 16);
  return __builtin_bit_cast(__bf16, s);
}

// one 16-byte chunk: global -> LDS
__device__ __forceinline__ void stage16(const __bf16* __restrict__ g,
                                        __bf16* l) {
#if HAS_ASYNC_LDS
  __builtin_amdgcn_global_load_async_to_lds_b128((gv4i*)g, (sv4i*)l, 0, 0);
#else
  *(v8bf*)l = *(const v8bf*)g;
#endif
}

__device__ __forceinline__ void wait_stage() {
#if HAS_ASYNC_LDS
#if __has_builtin(__builtin_amdgcn_s_wait_asynccnt)
  __builtin_amdgcn_s_wait_asynccnt(0);
#else
  asm volatile("s_wait_asynccnt 0" ::: "memory");
#endif
#endif
}

__global__ void cast_f32_to_bf16_k(const float* __restrict__ in,
                                   __bf16* __restrict__ out, int n) {
  int i = blockIdx.x * blockDim.x + threadIdx.x;
  if (i < n) out[i] = f2bf(in[i]);
}

// in: [R x C] f32 (row-major) -> out: [C x R] bf16 (row-major). R,C % 32 == 0.
__global__ void transpose_cast_k(const float* __restrict__ in,
                                 __bf16* __restrict__ out, int R, int C) {
  __shared__ float t[32][33];
  const int bc = blockIdx.x * 32, br = blockIdx.y * 32;
  const int tx = threadIdx.x & 31, ty = threadIdx.x >> 5;  // 256 thr: 8 rows/pass
#pragma unroll
  for (int i = 0; i < 32; i += 8)
    t[ty + i][tx] = in[(size_t)(br + ty + i) * C + (bc + tx)];
  __syncthreads();
#pragma unroll
  for (int i = 0; i < 32; i += 8)
    out[(size_t)(bc + ty + i) * R + (br + tx)] = f2bf(t[tx][ty + i]);
}

// EPI: 0 = bf16 out (scores/attn), 1 = f32 out + bias (final), 2 = qkv split.
// NT GEMM: A [M x K] row-major, B [N x K] row-major, C = A * B^T.
template <int EPI>
__launch_bounds__(256)
__global__ void gemm_wmma_nt(const __bf16* __restrict__ Ag,
                             const __bf16* __restrict__ Bg,
                             const float* __restrict__ bias, float scale,
                             float* __restrict__ outF, __bf16* __restrict__ outH,
                             __bf16* __restrict__ Qo, __bf16* __restrict__ Ko,
                             __bf16* __restrict__ Vo,
                             int M, int N, int Kd, int batchB) {
  __shared__ __bf16 lA[2 * ASZ];
  __shared__ __bf16 lB[2 * BSZ];

  const int tid  = threadIdx.x;
  const int lane = tid & 31;
  const int wave = tid >> 5;
  const int l16  = lane & 15;
  const int bm   = blockIdx.y * BLK_M;
  const int bn   = blockIdx.x * BLK_N;
  const int bz   = blockIdx.z;

  const __bf16* A = Ag + (size_t)bz * (size_t)M * (size_t)Kd;
  const __bf16* B = Bg + (batchB ? (size_t)bz * (size_t)N * (size_t)Kd : (size_t)0);

  // 8 waves: 2 (M) x 4 (N); each wave owns 64x64 = 4x4 wmma tiles
  const int wm = (wave >> 2) * 64;
  const int wn = (wave & 3) * 64;
  const int ah = (lane < 16) ? 0 : 8;    // ISA 16-bit A 16x32 per-lane K offsets
  const int bh = (lane < 16) ? 0 : 8;    // ISA 16-bit B 32x16 per-lane K offsets (same as A)

  v8f acc[4][4];
#pragma unroll
  for (int i = 0; i < 4; ++i)
#pragma unroll
    for (int j = 0; j < 4; ++j)
      acc[i][j] = v8f{0.f, 0.f, 0.f, 0.f, 0.f, 0.f, 0.f, 0.f};

  const int sr = tid >> 2;        // 0..63
  const int sc = (tid & 3) * 8;   // 0,8,16,24

  auto stageA = [&](int k0, __bf16* dst) {
#pragma unroll
    for (int p = 0; p < 2; ++p) {
      int r = sr + p * 64;
      stage16(A + (size_t)(bm + r) * Kd + (k0 + sc), &dst[r * LDSP + sc]);
    }
  };
  auto stageB = [&](int k0, __bf16* dst) {
#pragma unroll
    for (int p = 0; p < 4; ++p) {
      int r = sr + p * 64;
      stage16(B + (size_t)(bn + r) * Kd + (k0 + sc), &dst[r * LDSP + sc]);
    }
  };

  const int nk = Kd / BLK_K;
  stageA(0, lA);
  stageB(0, lB);
  wait_stage();
  __syncthreads();

  int buf = 0;
  for (int kt = 0; kt < nk; ++kt) {
    const __bf16* cA = lA + buf * ASZ;
    const __bf16* cB = lB + buf * BSZ;

    if (kt + 1 < nk) {               // fill the other buffer while we compute
      stageA((kt + 1) * BLK_K, lA + (buf ^ 1) * ASZ);
      stageB((kt + 1) * BLK_K, lB + (buf ^ 1) * BSZ);
    }
    if (kt + 2 < nk) {               // pull k+2 into L2 (global_prefetch_b8)
      const int kp = (kt + 2) * BLK_K;
      __builtin_prefetch(A + (size_t)(bm + sr) * Kd + (kp + sc), 0, 1);
      __builtin_prefetch(B + (size_t)(bn + sr) * Kd + (kp + sc), 0, 1);
      __builtin_prefetch(B + (size_t)(bn + 64 + sr) * Kd + (kp + sc), 0, 1);
    }

    // fragments in the exact ISA per-lane layouts
    v16bf afr[4], bfr[4];
#pragma unroll
    for (int i = 0; i < 4; ++i) {
      int m = wm + i * 16 + l16;
      v8bf lo = *(const v8bf*)&cA[m * LDSP + ah];
      v8bf hi = *(const v8bf*)&cA[m * LDSP + ah + 16];
      afr[i] = __builtin_shufflevector(lo, hi, 0, 1, 2, 3, 4, 5, 6, 7, 8, 9, 10,
                                       11, 12, 13, 14, 15);
    }
#pragma unroll
    for (int j = 0; j < 4; ++j) {
      int n = wn + j * 16 + l16;
      v8bf lo = *(const v8bf*)&cB[n * LDSP + bh];
      v8bf hi = *(const v8bf*)&cB[n * LDSP + bh + 16];
      bfr[j] = __builtin_shufflevector(lo, hi, 0, 1, 2, 3, 4, 5, 6, 7, 8, 9, 10,
                                       11, 12, 13, 14, 15);
    }

#pragma unroll
    for (int i = 0; i < 4; ++i)
#pragma unroll
      for (int j = 0; j < 4; ++j)
        acc[i][j] = __builtin_amdgcn_wmma_f32_16x16x32_bf16(
            false, afr[i], false, bfr[j], (short)0, acc[i][j], false, false);

    wait_stage();      // our async DMAs into buf^1 have landed in LDS
    __syncthreads();   // everyone done reading buf -> safe to swap
    buf ^= 1;
  }

  // ---- epilogue ----
  const int rbase = (lane < 16) ? 0 : 8;   // C layout: VGPR r -> M = r (+8 hi half)
#pragma unroll
  for (int j = 0; j < 4; ++j) {
    const int ncol = bn + wn + j * 16 + l16;
    const float bv = (EPI == 0) ? 0.0f : bias[ncol];
#pragma unroll
    for (int i = 0; i < 4; ++i) {
#pragma unroll
      for (int r = 0; r < 8; ++r) {
        const int mrow = bm + wm + i * 16 + rbase + r;
        const float val = acc[i][j][r] * scale + bv;
        if (EPI == 0) {
          outH[(size_t)bz * (size_t)M * N + (size_t)mrow * N + ncol] = f2bf(val);
        } else if (EPI == 1) {
          outF[(size_t)mrow * N + ncol] = val;
        } else {
          // de-interleave: qkv is the fastest-varying split axis (col = 3*d + s)
          const int s = ncol % 3, col = ncol / 3;
          __bf16* dst = (s == 0) ? Qo : (s == 1) ? Ko : Vo;
          dst[(size_t)mrow * 2048 + col] = f2bf(val);  // mrow already = b*2048+seq
        }
      }
    }
  }
}

extern "C" void kernel_launch(void* const* d_in, const int* in_sizes, int n_in,
                              void* d_out, int out_size, void* d_ws, size_t ws_size,
                              hipStream_t stream) {
  (void)in_sizes; (void)n_in; (void)out_size; (void)ws_size;
  constexpr int BATCH = 4, SEQ = 2048, DM = 2048;
  constexpr int MFLAT = BATCH * SEQ;                 // 8192
  constexpr int N3 = 3 * DM;                         // 6144
  constexpr float SCALING = 0.022097086912079608f;   // 2048^-0.5

  const float* x     = (const float*)d_in[0];
  const float* W_qkv = (const float*)d_in[1];
  const float* b_qkv = (const float*)d_in[2];
  const float* W_fc  = (const float*)d_in[3];
  const float* b_fc  = (const float*)d_in[4];
  float* out = (float*)d_out;

  // workspace carve-up (bf16 buffers), 256B aligned
  char* base = (char*)d_ws;
  size_t off = 0;
  auto take = [&](size_t elems) {
    off = (off + 255) & ~(size_t)255;
    char* p = base + off;
    off += elems * sizeof(__bf16);
    return (__bf16*)p;
  };
  __bf16* xb  = take((size_t)MFLAT * DM);
  __bf16* wqT = take((size_t)N3 * DM);     // W_qkv^T  [6144 x 2048]
  __bf16* wfT = take((size_t)DM * DM);     // W_fc^T   [2048 x 2048]
  __bf16* Q   = take((size_t)MFLAT * DM);
  __bf16* K   = take((size_t)MFLAT * DM);
  __bf16* V   = take((size_t)MFLAT * DM);
  __bf16* S   = take((size_t)MFLAT * SEQ); // scores, per-batch 2048x2048
  __bf16* Atn = take((size_t)MFLAT * DM);

  cast_f32_to_bf16_k<<<(MFLAT * DM + 255) / 256, 256, 0, stream>>>(x, xb, MFLAT * DM);
  transpose_cast_k<<<dim3(N3 / 32, DM / 32), 256, 0, stream>>>(W_qkv, wqT, DM, N3);
  transpose_cast_k<<<dim3(DM / 32, DM / 32), 256, 0, stream>>>(W_fc, wfT, DM, DM);

  // 1) qkv = x @ W_qkv + b_qkv, split-scattered into Q/K/V (M=8192 N=6144 K=2048)
  gemm_wmma_nt<2><<<dim3(N3 / BLK_N, MFLAT / BLK_M, 1), 256, 0, stream>>>(
      xb, wqT, b_qkv, 1.0f, nullptr, nullptr, Q, K, V, MFLAT, N3, DM, 0);

  // 2) scores = (Q @ K^T) * d^-1/2   (per batch 2048^3)
  gemm_wmma_nt<0><<<dim3(SEQ / BLK_N, SEQ / BLK_M, BATCH), 256, 0, stream>>>(
      Q, K, nullptr, SCALING, nullptr, S, nullptr, nullptr, nullptr, SEQ, SEQ, DM, 1);

  // 3) attn = scores @ V^T           (per batch 2048^3)
  gemm_wmma_nt<0><<<dim3(SEQ / BLK_N, SEQ / BLK_M, BATCH), 256, 0, stream>>>(
      S, V, nullptr, 1.0f, nullptr, Atn, nullptr, nullptr, nullptr, SEQ, SEQ, SEQ, 1);

  // 4) out = attn @ W_fc + b_fc      (M=8192 N=2048 K=2048, f32 out)
  gemm_wmma_nt<1><<<dim3(DM / BLK_N, MFLAT / BLK_M, 1), 256, 0, stream>>>(
      Atn, wfT, b_fc, 1.0f, out, nullptr, nullptr, nullptr, nullptr, MFLAT, DM, DM, 0);
}
